// Recurrent_83975200571955
// MI455X (gfx1250) — matrix operand order repaired and checked by reading, not compiled, not run on hardware
//
#include <hip/hip_runtime.h>

#define BATCH 128
#define SEQ   500
#define DIM   128   // DT = DH = U = 128
#define MDIM  32
#define BS    8     // batch rows per recurrent workgroup
#define NWG   (BATCH / BS)

typedef __attribute__((ext_vector_type(16))) _Float16 v16h;
typedef __attribute__((ext_vector_type(8)))  _Float16 v8h;
typedef __attribute__((ext_vector_type(8)))  float    v8f;

// Fragment buffer offsets (in halves) inside workspace
#define OFF_WIN    0          // K=256 -> 8 chunks * 8 tiles * 32 lanes * 16 = 32768
#define OFF_WOUT   32768      // K=128 -> 16384
#define OFF_WTIME  49152      // K=256 (row-permuted) -> 32768
#define OFF_WATT   81920
#define OFF_WHINT  114688
#define OFF_WGATE  147456     // K=384 -> 49152
#define FRAG_TOTAL 196608
#define OFF_ACTS   FRAG_TOTAL // acts f16: B*S*DIM halves

__device__ __forceinline__ v8f wmma16x16x32(v16h a, v16h b, v8f c) {
  // D = A(16x32 f16) * B(32x16 f16) + C(16x16 f32)
  return __builtin_amdgcn_wmma_f32_16x16x32_f16(false, a, false, b, (short)0, c,
                                                false, false);
}

// A-fragment (16x32 f16) from an LDS row, per ISA layout:
// lane L (m = L&15, hi = L>>4): halves 0..7  = A[m][32c + 8*hi + 0..7]
//                               halves 8..15 = A[m][32c + 16 + 8*hi + 0..7]
__device__ __forceinline__ v16h load_afrag(const _Float16* __restrict__ Xrow,
                                           int c, int hi) {
  const _Float16* p = Xrow + 32 * c + 8 * hi;
  v8h lo = *(const v8h*)p;
  v8h hh = *(const v8h*)(p + 16);
  v16h a;
#pragma unroll
  for (int i = 0; i < 8; ++i) { a[i] = lo[i]; a[8 + i] = hh[i]; }
  return a;
}

// B-fragment (32x16 f16): pre-packed so each lane reads 32 contiguous bytes.
__device__ __forceinline__ v16h load_bfrag(const _Float16* __restrict__ frag,
                                           int c, int j, int lane) {
  return *(const v16h*)(frag + ((((c << 3) + j) << 5) + lane) * 16);
}

__device__ __forceinline__ float sigmoid_f(float x) {
  return 1.f / (1.f + __expf(-x));
}
__device__ __forceinline__ float tanh_f(float x) {
  x = fminf(fmaxf(x, -15.f), 15.f);
  float e2 = __expf(2.f * x);
  return (e2 - 1.f) / (e2 + 1.f);
}

// ---------------------------------------------------------------------------
// Kernel 0: convert weights f32 -> f16 in WMMA B-fragment order.
// Fragment flat index: (((c*8 + j)*32 + lane)*16 + h)
//   k = 32*c + 8*(lane>>4) + (h&7) + 16*(h>>3) ; n = 16*j + (lane&15)
// Gain matrices (W_time/att/hint, shape 257x128): staged-K 0..127 = h_tilde
// (src row 129+k), staged-K 128..255 = acts (src row k-127). Row 0 (fac) is
// applied separately as a rank-1 add.
// ---------------------------------------------------------------------------
__global__ __launch_bounds__(256) void prep_frags(
    const float* __restrict__ W_in,  const float* __restrict__ W_out,
    const float* __restrict__ W_time, const float* __restrict__ W_att,
    const float* __restrict__ W_hint, const float* __restrict__ W_gate,
    _Float16* __restrict__ frag) {
  int fid = blockIdx.x * 256 + threadIdx.x;
  if (fid >= FRAG_TOTAL) return;
  const float* src; int base; int perm = 0;
  if (fid < OFF_WOUT)       { src = W_in;   base = OFF_WIN; }
  else if (fid < OFF_WTIME) { src = W_out;  base = OFF_WOUT; }
  else if (fid < OFF_WATT)  { src = W_time; base = OFF_WTIME; perm = 1; }
  else if (fid < OFF_WHINT) { src = W_att;  base = OFF_WATT;  perm = 1; }
  else if (fid < OFF_WGATE) { src = W_hint; base = OFF_WHINT; perm = 1; }
  else                      { src = W_gate; base = OFF_WGATE; }
  int local = fid - base;
  int h    = local & 15;
  int lane = (local >> 4) & 31;
  int j    = (local >> 9) & 7;
  int c    = local >> 12;
  int k    = 32 * c + 8 * (lane >> 4) + (h & 7) + ((h >> 3) << 4);
  int n    = 16 * j + (lane & 15);
  int srck = perm ? ((k < 128) ? (129 + k) : (k - 127)) : k;
  frag[fid] = (_Float16)src[srck * DIM + n];
}

// ---------------------------------------------------------------------------
// Kernel 1: acts = relu([emb_topic[t], emb_resps[r]] @ W_in + b_in) -> f16
// One 16-row tile per block; 8 waves, each owns a 16x16 output tile; K=256.
// ---------------------------------------------------------------------------
__global__ __launch_bounds__(256) void acts_kernel(
    const int* __restrict__ topics, const int* __restrict__ resps,
    const float* __restrict__ emb_topic, const float* __restrict__ emb_resps,
    const float* __restrict__ b_in,
    const _Float16* __restrict__ fragWin, _Float16* __restrict__ acts) {
  __shared__ __align__(16) _Float16 sA[16][256];
  int tid = threadIdx.x;
  int tile = blockIdx.x;
#pragma unroll
  for (int q = 0; q < 16; ++q) {
    int hidx = tid + 256 * q;
    int r = hidx >> 8, c = hidx & 255;
    int g = tile * 16 + r;
    float v = (c < 128) ? emb_topic[topics[g] * 128 + c]
                        : emb_resps[resps[g] * 128 + (c - 128)];
    sA[r][c] = (_Float16)v;
  }
  __syncthreads();
  int lane = tid & 31, j = tid >> 5;
  int m = lane & 15, hi = lane >> 4;
  v8f R = {};
#pragma unroll
  for (int c = 0; c < 8; ++c) {
    v16h a = load_afrag(&sA[m][0], c, hi);
    v16h b = load_bfrag(fragWin, c, j, lane);
    R = wmma16x16x32(a, b, R);
  }
  int u = 16 * j + m;
  float bi = b_in[u];
#pragma unroll
  for (int v = 0; v < 8; ++v) {
    int row = v + 8 * hi;                 // C/D layout: VGPR v holds M = v + 8*hi
    int g = tile * 16 + row;
    float val = R[v] + bi;
    acts[g * 128 + u] = (_Float16)(val > 0.f ? val : 0.f);
  }
}

// ---------------------------------------------------------------------------
// Kernel 2: 500-step recurrence. 16 WGs x 256 threads; 8 batch rows per WG.
// H (8x32x128 f32) lives in LDS. A-staging X = [h_tilde | acts | topic_emb]
// (16 rows, rows 8..15 zero padding). 40 WMMAs per wave per step.
// ---------------------------------------------------------------------------
__global__ __launch_bounds__(256) void recurrent_kernel(
    const int* __restrict__ topics,
    const float* __restrict__ time_factor, const float* __restrict__ attempt_factor,
    const float* __restrict__ hint_factor, const int* __restrict__ masks,
    const float* __restrict__ emb_topic, const float* __restrict__ q_matrix,
    const float* __restrict__ init_h,
    const float* __restrict__ b_out,
    const float* __restrict__ W_time, const float* __restrict__ b_time,
    const float* __restrict__ W_att,  const float* __restrict__ b_att,
    const float* __restrict__ W_hint, const float* __restrict__ b_hint,
    const float* __restrict__ W_cap,  const float* __restrict__ b_cap,
    const float* __restrict__ w_lg,   const float* __restrict__ b_gate,
    const _Float16* __restrict__ frag, const _Float16* __restrict__ acts,
    float* __restrict__ outPred, float* __restrict__ outImp) {
  __shared__ __align__(16) float    sH[BS][MDIM][DIM];   // 128 KB
  __shared__ __align__(16) _Float16 sX[16][384];         // 12 KB
  __shared__ float sLG[BS][DIM];
  __shared__ float sGam[BS][DIM];
  __shared__ float sSig[BS][DIM];
  __shared__ float sW[BS][MDIM];
  __shared__ float sBout[DIM], sBtime[DIM], sBatt[DIM], sBhint[DIM], sBgate[DIM];
  __shared__ float sW0t[DIM], sW0a[DIM], sW0h[DIM];
  __shared__ float sFt[BS], sFa[BS], sFh[BS], sCap[BS];
  __shared__ int   sMk[BS], sTI[BS];
  __shared__ float sWcap[8], sWlg[3], sBcap;
  __shared__ float sPS[BS][8], sPL[BS][8];

  const _Float16* fragWout  = frag + OFF_WOUT;
  const _Float16* fragWtime = frag + OFF_WTIME;
  const _Float16* fragWatt  = frag + OFF_WATT;
  const _Float16* fragWhint = frag + OFF_WHINT;
  const _Float16* fragWgate = frag + OFF_WGATE;

  int tid = threadIdx.x;
  int wg  = blockIdx.x;

  // ---- init ----
#pragma unroll 1
  for (int q = 0; q < 128; ++q) {               // H0 = broadcast(init_h)
    int idx = tid + 256 * q;
    int b = idx >> 12;
    int rem = idx & 4095;                       // m*128 + u
    sH[b][rem >> 7][rem & 127] = init_h[rem];
  }
  {
    _Float16* xp = &sX[0][0];
#pragma unroll
    for (int q = 0; q < 24; ++q) xp[tid + 256 * q] = (_Float16)0.f;
  }
  if (tid < 128) {
    sBout[tid] = b_out[tid];  sBtime[tid] = b_time[tid];
    sBatt[tid] = b_att[tid];  sBhint[tid] = b_hint[tid];
    sBgate[tid] = b_gate[tid];
    sW0t[tid] = W_time[tid];  sW0a[tid] = W_att[tid];  sW0h[tid] = W_hint[tid];
  }
  if (tid < 8)  sWcap[tid] = W_cap[tid];
  if (tid < 3)  sWlg[tid]  = w_lg[tid];
  if (tid == 0) sBcap = b_cap[0];
  __syncthreads();

  int lane = tid & 31, j = tid >> 5;
  int ml = lane & 15, hi = lane >> 4;

  for (int s = 0; s < SEQ; ++s) {
    // ---- phase A: scalars + gathers ----
    if (tid < BS) {
      int b = tid, bb = wg * BS + b, idx = bb * SEQ + s;
      sTI[b] = topics[idx];
      float t = time_factor[idx], at = attempt_factor[idx], hn = hint_factor[idx];
      sFt[b] = t; sFa[b] = at; sFh[b] = hn;
      sMk[b] = masks[idx];
      float cs = t * sWcap[0] + at * sWcap[1] + hn * sWcap[2] + t * at * sWcap[3] +
                 t * hn * sWcap[4] + at * hn * sWcap[5] + t * at * hn * sWcap[6] +
                 sWcap[7] + sBcap;
      sCap[b] = sigmoid_f(cs);
    }
    __syncthreads();
    {
      int b = tid >> 5, m = tid & 31;           // corr gather: q_matrix[topic]
      sW[b][m] = q_matrix[sTI[b] * MDIM + m];
    }
#pragma unroll
    for (int q = 0; q < 4; ++q) {
      int p = tid + 256 * q;
      int b = p >> 7, u = p & 127;
      int bb = wg * BS + b;
      sX[b][128 + u] = acts[(bb * SEQ + s) * 128 + u];
      sX[b][256 + u] = (_Float16)emb_topic[sTI[b] * 128 + u];
    }
    __syncthreads();

    // ---- phase B: h_tilde[b][u] = sum_m w[b][m] * H[b][m][u] ----
#pragma unroll
    for (int q = 0; q < 4; ++q) {
      int p = tid + 256 * q;
      int b = p >> 7, u = p & 127;
      float acc = 0.f;
#pragma unroll
      for (int m = 0; m < MDIM; ++m) acc += sW[b][m] * sH[b][m][u];
      sX[b][u] = (_Float16)acc;
    }
    __syncthreads();

    // ---- phase C: five GEMMs via WMMA, shared A fragments ----
    v8f Ro = {}, Rt = {}, Ra = {}, Rh = {}, Rg = {};
    const _Float16* Xrow = &sX[ml][0];
#pragma unroll
    for (int c = 0; c < 12; ++c) {
      v16h a = load_afrag(Xrow, c, hi);
      if (c < 4)
        Ro = wmma16x16x32(a, load_bfrag(fragWout, c, j, lane), Ro);
      if (c < 8) {
        Rt = wmma16x16x32(a, load_bfrag(fragWtime, c, j, lane), Rt);
        Ra = wmma16x16x32(a, load_bfrag(fragWatt,  c, j, lane), Ra);
        Rh = wmma16x16x32(a, load_bfrag(fragWhint, c, j, lane), Rh);
      }
      Rg = wmma16x16x32(a, load_bfrag(fragWgate, c, j, lane), Rg);
    }
    if (hi == 0) {                               // lanes 0-15 hold rows 0..7
      int u = 16 * j + ml;
      float bo = sBout[u], bt = sBtime[u], ba = sBatt[u], bh = sBhint[u],
            bg = sBgate[u];
      float w0t = sW0t[u], w0a = sW0a[u], w0h = sW0h[u];
      float wl0 = sWlg[0], wl1 = sWlg[1], wl2 = sWlg[2];
#pragma unroll
      for (int v = 0; v < 8; ++v) {
        int b = v;
        float ft = sFt[b], fa = sFa[b], fh = sFh[b];
        sSig[b][u] = sigmoid_f(Ro[v] + bo);
        float gt = tanh_f(Rt[v] + ft * w0t + bt) * ft;
        float ga = tanh_f(Ra[v] + fa * w0a + ba) * fa;
        float gh = tanh_f(Rh[v] + fh * w0h + bh) * fh;
        float pre = wl0 * gt + wl1 * ga + wl2 * gh;
        sLG[b][u]  = sCap[b] * (pre > 0.f ? pre : 0.f);
        sGam[b][u] = sigmoid_f(Rg[v] + bg);
      }
    }
    __syncthreads();

    // ---- phase D1: partial reductions over u ----
    if (tid < 64) {
      int b = tid >> 3, seg = tid & 7;
      float s1 = 0.f;
#pragma unroll
      for (int i = 0; i < 16; ++i) s1 += sSig[b][seg * 16 + i];
      sPS[b][seg] = s1;
    } else if (tid < 128) {
      int t2 = tid - 64;
      int b = t2 >> 3, seg = t2 & 7;
      float s1 = 0.f;
#pragma unroll
      for (int i = 0; i < 16; ++i) s1 += sLG[b][seg * 16 + i];
      sPL[b][seg] = s1;
    }
    __syncthreads();

    // ---- phase D2: outputs + masked H update ----
    if (tid < BS) {
      int b = tid;
      float ps = 0.f, is = 0.f;
#pragma unroll
      for (int k2 = 0; k2 < 8; ++k2) { ps += sPS[b][k2]; is += sPL[b][k2]; }
      ps *= (1.f / 128.f); is *= (1.f / 128.f);
      float pv = sMk[b] ? ps : 0.f;
      float iv = sMk[b] ? is : 0.f;
      int bb = wg * BS + b;
      if (s >= 1) outPred[bb * (SEQ - 1) + (s - 1)] = pv;   // preds[:, 1:]
      outImp[bb * SEQ + s] = iv;
    }
#pragma unroll
    for (int q = 0; q < 4; ++q) {
      int p = tid + 256 * q;
      int b = p >> 7, u = p & 127;
      if (sMk[b]) {
        float gm = sGam[b][u], lg = sLG[b][u];
#pragma unroll
        for (int m = 0; m < MDIM; ++m)
          sH[b][m][u] = gm * sH[b][m][u] + sW[b][m] * lg;
      }
    }
    __syncthreads();
  }
}

// ---------------------------------------------------------------------------
extern "C" void kernel_launch(void* const* d_in, const int* in_sizes, int n_in,
                              void* d_out, int out_size, void* d_ws, size_t ws_size,
                              hipStream_t stream) {
  (void)in_sizes; (void)n_in; (void)out_size; (void)ws_size;
  const int*   topics         = (const int*)d_in[0];
  const int*   resps          = (const int*)d_in[1];
  const float* time_factor    = (const float*)d_in[2];
  const float* attempt_factor = (const float*)d_in[3];
  const float* hint_factor    = (const float*)d_in[4];
  const int*   masks          = (const int*)d_in[5];
  // d_in[6] = training (unused)
  const float* emb_topic = (const float*)d_in[7];
  const float* emb_resps = (const float*)d_in[8];
  const float* q_matrix  = (const float*)d_in[9];
  const float* W_in   = (const float*)d_in[10];
  const float* b_in   = (const float*)d_in[11];
  const float* init_h = (const float*)d_in[12];
  const float* W_out  = (const float*)d_in[13];
  const float* b_out  = (const float*)d_in[14];
  const float* W_time = (const float*)d_in[15];
  const float* b_time = (const float*)d_in[16];
  const float* W_att  = (const float*)d_in[17];
  const float* b_att  = (const float*)d_in[18];
  const float* W_hint = (const float*)d_in[19];
  const float* b_hint = (const float*)d_in[20];
  const float* W_cap  = (const float*)d_in[21];
  const float* b_cap  = (const float*)d_in[22];
  const float* w_lg   = (const float*)d_in[23];
  const float* W_gate = (const float*)d_in[24];
  const float* b_gate = (const float*)d_in[25];

  _Float16* frag    = (_Float16*)d_ws;
  _Float16* actsBuf = frag + OFF_ACTS;
  float* outPred = (float*)d_out;
  float* outImp  = outPred + BATCH * (SEQ - 1);

  prep_frags<<<FRAG_TOTAL / 256, 256, 0, stream>>>(W_in, W_out, W_time, W_att,
                                                   W_hint, W_gate, frag);
  acts_kernel<<<(BATCH * SEQ) / 16, 256, 0, stream>>>(
      topics, resps, emb_topic, emb_resps, b_in, frag + OFF_WIN, actsBuf);
  recurrent_kernel<<<NWG, 256, 0, stream>>>(
      topics, time_factor, attempt_factor, hint_factor, masks, emb_topic,
      q_matrix, init_h, b_out, W_time, b_time, W_att, b_att, W_hint, b_hint,
      W_cap, b_cap, w_lg, b_gate, frag, actsBuf, outPred, outImp);
}
